// TransformerBlock_79611513799236
// MI455X (gfx1250) — matrix operand order repaired and checked by reading, not compiled, not run on hardware
//
#include <hip/hip_runtime.h>

#define EMB   1024
#define HEADS 16
#define HDIM  64
#define FF    4096
#define SEQ   2048
#define BATCH 2
#define MTOT  (BATCH * SEQ)   // 4096 rows

typedef __attribute__((ext_vector_type(16))) __bf16 v16bf;
typedef __attribute__((ext_vector_type(8)))  float  v8f;
typedef __attribute__((ext_vector_type(4))) unsigned u32x4;
typedef __attribute__((ext_vector_type(8))) int      i32x8;
typedef __attribute__((ext_vector_type(4))) int      i32x4;

union BFrag {
  v16bf v;
  unsigned u[8];
  unsigned short s[16];
};

__device__ inline unsigned short f2bf(float f) {
  union { float f; unsigned u; } x; x.f = f;
  unsigned r = x.u + 0x7FFFu + ((x.u >> 16) & 1u);   // round-to-nearest-even
  return (unsigned short)(r >> 16);
}

__device__ inline float gelu_tanh(float x) {
  return 0.5f * x * (1.0f + tanhf(0.7978845608028654f * (x + 0.044715f * x * x * x)));
}

// ---------------------------------------------------------------- TDM 2-D tile load
// D# per CDNA5 ISA: group0 = {flags, lds_addr, global_addr, type=2},
// group1 = {data_size=2B, tensor dims, tile dims, dim0 stride}. Groups 2/3 zero (2-D).
// This toolchain exposes the 6-arg builtin: (g0, g1, g2, g3, g4, cpol).
__device__ inline void tdm_load_tile_bf16(unsigned lds_addr, const void* gaddr,
                                          unsigned tensor_d0, unsigned tensor_d1,
                                          unsigned tile_d0, unsigned tile_d1,
                                          unsigned long long stride0_elems) {
  unsigned long long ga = (unsigned long long)gaddr;
  u32x4 g0;
  g0[0] = 1u;                                            // count=1, user descriptor
  g0[1] = lds_addr;                                      // LDS byte address
  g0[2] = (unsigned)(ga & 0xFFFFFFFFu);                  // global_addr[31:0]
  g0[3] = (unsigned)((ga >> 32) & 0x01FFFFFFu) | (2u << 30);  // addr[56:32] | type=2
  i32x8 g1;
  g1[0] = (int)(1u << 16);                               // wg_mask=0, data_size=1 (2 bytes)
  g1[1] = (int)((tensor_d0 & 0xFFFFu) << 16);            // tensor_dim0[15:0] @bits63:48
  g1[2] = (int)(((tensor_d0 >> 16) & 0xFFFFu) | ((tensor_d1 & 0xFFFFu) << 16));
  g1[3] = (int)(((tensor_d1 >> 16) & 0xFFFFu) | ((tile_d0 & 0xFFFFu) << 16));
  g1[4] = (int)(tile_d1 & 0xFFFFu);                      // tile_dim1; tile_dim2=0
  g1[5] = (int)(stride0_elems & 0xFFFFFFFFu);            // tensor_dim0_stride[31:0]
  g1[6] = (int)((stride0_elems >> 32) & 0xFFFFu);        // stride[47:32]; dim1_stride=0
  g1[7] = 0;
  i32x4 z4 = {0, 0, 0, 0};
  i32x8 z8 = {0, 0, 0, 0, 0, 0, 0, 0};
  __builtin_amdgcn_tensor_load_to_lds(g0, g1, z4, z4, z8, 0);
}

__device__ inline unsigned lds_addr_of(const void* p) {
  // flat LDS address = {SHARED_BASE, offset32}; low 32 bits are the LDS byte offset
  return (unsigned)(unsigned long long)(size_t)p;
}

// ---------------------------------------------------------------- fp32 -> bf16 (plain)
__global__ void cvt_f32_bf16(const float* __restrict__ in,
                             unsigned short* __restrict__ out, int n) {
  int i = blockIdx.x * blockDim.x + threadIdx.x;
  int stride = gridDim.x * blockDim.x;
  for (; i < n; i += stride) out[i] = f2bf(in[i]);
}

// ---------------------------------------------------------------- fp32 KxN -> bf16 NxK (transpose)
__global__ __launch_bounds__(256) void cvt_transpose_bf16(
    const float* __restrict__ in, unsigned short* __restrict__ out, int K, int N) {
  __shared__ float tile[32][33];
  int k0 = blockIdx.y * 32, n0 = blockIdx.x * 32;
  int tx = threadIdx.x & 31, ty = threadIdx.x >> 5;   // 8 row-groups
#pragma unroll
  for (int r = ty; r < 32; r += 8)
    tile[r][tx] = in[(size_t)(k0 + r) * N + n0 + tx];
  __syncthreads();
#pragma unroll
  for (int r = ty; r < 32; r += 8)
    out[(size_t)(n0 + r) * K + k0 + tx] = f2bf(tile[tx][r]);
}

// ---------------------------------------------------------------- LayerNorm (row=1024) -> bf16
__global__ __launch_bounds__(256) void layernorm_bf16(
    const float* __restrict__ x, const float* __restrict__ g,
    const float* __restrict__ s, unsigned short* __restrict__ out) {
  __shared__ float red[18];
  int row = blockIdx.x;
  int tid = threadIdx.x;
  size_t base = (size_t)row * EMB;

  float t0 = x[base + tid], t1 = x[base + tid + 256];
  float t2 = x[base + tid + 512], t3 = x[base + tid + 768];
  float sum = t0 + t1 + t2 + t3;
  float sq  = t0 * t0 + t1 * t1 + t2 * t2 + t3 * t3;
#pragma unroll
  for (int off = 16; off >= 1; off >>= 1) {
    sum += __shfl_xor(sum, off, 32);
    sq  += __shfl_xor(sq,  off, 32);
  }
  int wave = tid >> 5, lane = tid & 31;
  if (lane == 0) { red[wave] = sum; red[8 + wave] = sq; }
  __syncthreads();
  if (tid == 0) {
    float S = 0.f, Q = 0.f;
    for (int w = 0; w < 8; ++w) { S += red[w]; Q += red[8 + w]; }
    float mean = S / (float)EMB;
    float var  = Q / (float)EMB - mean * mean;
    red[16] = mean;
    red[17] = rsqrtf(var + 1e-5f);
  }
  __syncthreads();
  float mean = red[16], rstd = red[17];
#pragma unroll
  for (int j = 0; j < 4; ++j) {
    int idx = tid + j * 256;
    float t = x[base + idx];
    out[base + idx] = f2bf(g[idx] * ((t - mean) * rstd) + s[idx]);
  }
}

// ---------------------------------------------------------------- WMMA GEMM (TDM-staged)
// C(MxN) = A(MxK bf16 row-major) * Bt^T, with Bt = B^T stored (N x K bf16 row-major).
// Block: 128 threads (4 waves, 2x2), block tile 128x128, BK=32, wave tile 64x64
// (16 wmma/stage/wave). A and Bt tiles are DMA'd by the Tensor Data Mover into
// ping-pong LDS buffers; one barrier per stage.
template <bool OUT_BF16, bool DO_GELU>
__global__ __launch_bounds__(128) void wmma_gemm_bf16(
    const unsigned short* __restrict__ A, const unsigned short* __restrict__ Bt,
    const float* __restrict__ bias, const float* __restrict__ residual,
    void* __restrict__ outp, int M, int N, int K) {
  __shared__ unsigned As[2][128 * 16];   // 128 rows x 32 bf16 (row-major), x2 buffers
  __shared__ unsigned Bs[2][128 * 16];

  int tid  = threadIdx.x;
  int lane = tid & 31, wave = tid >> 5;
  int wr = wave >> 1, wc = wave & 1;
  int half = lane >> 4, lr = lane & 15;
  int m0 = blockIdx.y * 128, n0 = blockIdx.x * 128;
  (void)M;

  v8f acc[4][4] = {};
  int nst = K >> 5;

  if (wave == 0) {   // stage 0 tiles
    tdm_load_tile_bf16(lds_addr_of(&As[0][0]), A  + (size_t)m0 * K, K, (unsigned)M, 32, 128, (unsigned long long)K);
    tdm_load_tile_bf16(lds_addr_of(&Bs[0][0]), Bt + (size_t)n0 * K, K, (unsigned)N, 32, 128, (unsigned long long)K);
  }

  for (int st = 0; st < nst; ++st) {
    int buf = st & 1;
    if (wave == 0) __builtin_amdgcn_s_wait_tensorcnt(0);
    __syncthreads();                      // stage `st` tiles visible to all waves
    if (wave == 0 && st + 1 < nst) {      // kick off next stage into the other buffer
      int k1 = (st + 1) << 5;
      tdm_load_tile_bf16(lds_addr_of(&As[buf ^ 1][0]), A  + (size_t)m0 * K + k1, K, (unsigned)M, 32, 128, (unsigned long long)K);
      tdm_load_tile_bf16(lds_addr_of(&Bs[buf ^ 1][0]), Bt + (size_t)n0 * K + k1, K, (unsigned)N, 32, 128, (unsigned long long)K);
    }

    BFrag a[4], b[4];
#pragma unroll
    for (int mb = 0; mb < 4; ++mb) {      // A layout: lanes0-15 V0-3=K0..7,V4-7=K16..23; lanes16-31 +8
      const unsigned* Ar = &As[buf][(wr * 64 + mb * 16 + lr) * 16];
#pragma unroll
      for (int c = 0; c < 8; ++c) {
        int idx = (c < 4) ? (half * 4 + c) : (8 + half * 4 + (c - 4));
        a[mb].u[c] = Ar[idx];
      }
    }
#pragma unroll
    for (int nb = 0; nb < 4; ++nb) {      // B layout: lanes0-15 V0-3=K0..7,V4-7=K16..23; lanes16-31 +8
      const unsigned* Br = &Bs[buf][(wc * 64 + nb * 16 + lr) * 16];
#pragma unroll
      for (int c = 0; c < 8; ++c) b[nb].u[c] = Br[(c < 4) ? (half * 4 + c) : (8 + half * 4 + (c - 4))];
    }
#pragma unroll
    for (int mb = 0; mb < 4; ++mb)
#pragma unroll
      for (int nb = 0; nb < 4; ++nb)
        acc[mb][nb] = __builtin_amdgcn_wmma_f32_16x16x32_bf16(
            false, a[mb].v, false, b[nb].v, (short)0, acc[mb][nb], false, false);
    // no trailing barrier: next-stage TDM/readers touch only the other buffer,
    // and the top-of-stage barrier orders buffer reuse two stages apart.
  }

#pragma unroll
  for (int mb = 0; mb < 4; ++mb) {
#pragma unroll
    for (int nb = 0; nb < 4; ++nb) {
#pragma unroll
      for (int r = 0; r < 8; ++r) {       // C layout: M = r + 8*(lane/16), N = lane%16
        int m = m0 + wr * 64 + mb * 16 + r + half * 8;
        int n = n0 + wc * 64 + nb * 16 + lr;
        float val = acc[mb][nb][r];
        if (bias) val += bias[n];
        if (DO_GELU) val = gelu_tanh(val);
        if (residual) val += residual[(size_t)m * N + n];
        if (OUT_BF16)
          ((unsigned short*)outp)[(size_t)m * N + n] = f2bf(val);
        else
          ((float*)outp)[(size_t)m * N + n] = val;
      }
    }
  }
}

// ---------------------------------------------------------------- flash attention (causal)
// q,k,v,ctx: bf16, layout (b, s, h, d) == (MTOT, 1024) row-major.
// Block: 128 threads (4 waves); each wave owns 16 query rows; key tiles of 32.
__global__ __launch_bounds__(128) void flash_attn_wmma(
    const unsigned short* __restrict__ q, const unsigned short* __restrict__ k,
    const unsigned short* __restrict__ v, unsigned short* __restrict__ ctx) {
  __shared__ unsigned pbuf[4 * 16 * 16];   // per wave: 16 rows x 32 bf16 probs

  int tid  = threadIdx.x;
  int lane = tid & 31, wave = tid >> 5;
  int half = lane >> 4, lr = lane & 15;
  int b = blockIdx.x >> 4, h = blockIdx.x & 15;
  int qbase = blockIdx.y * 64;
  int qrow0 = qbase + wave * 16;

  const unsigned* q32 = (const unsigned*)q;
  const unsigned* k32 = (const unsigned*)k;

  // Q fragments (A operand), d 0..31 and 32..63
  BFrag a0, a1;
  {
    int s = qrow0 + lr;
    size_t base = ((size_t)(b * SEQ + s) * HEADS + h) * (HDIM / 2);
#pragma unroll
    for (int c = 0; c < 8; ++c) {
      int pidx = (c < 4) ? (half * 4 + c) : (8 + half * 4 + (c - 4));
      a0.u[c] = q32[base + pidx];
      a1.u[c] = q32[base + 16 + pidx];
    }
  }

  float m_i[8], l_i[8];
  v8f o[4] = {};
#pragma unroll
  for (int r = 0; r < 8; ++r) { m_i[r] = -1e30f; l_i[r] = 0.f; }

  int ktend = qbase + 64;                       // uniform across block
  for (int kt0 = 0; kt0 < ktend; kt0 += 32) {
    bool active = (kt0 <= qrow0 + 15);          // uniform within wave
    if (active) {
      if (kt0 + 32 < ktend) {                   // prefetch next key tile (global_prefetch_b8)
        int pkey = kt0 + 32 + half * 16 + lr;
        size_t pb = ((size_t)(b * SEQ + pkey) * HEADS + h) * HDIM;
        __builtin_prefetch(k + pb, 0, 0);
        __builtin_prefetch(v + pb, 0, 0);
      }
      v8f cs[2] = {};
#pragma unroll
      for (int nb = 0; nb < 2; ++nb) {          // scores = Q (16x64) . K^T (64x16)
        int key = kt0 + nb * 16 + lr;
        size_t kb = ((size_t)(b * SEQ + key) * HEADS + h) * (HDIM / 2);
        BFrag blo, bhi;
#pragma unroll
        for (int c = 0; c < 8; ++c) {
          blo.u[c] = k32[kb + ((c < 4) ? (half * 4 + c) : (8 + half * 4 + (c - 4)))];        // d = table order (8*half + 2c, 16 + 8*half + 2(c-4)) {+1}
          bhi.u[c] = k32[kb + 16 + ((c < 4) ? (half * 4 + c) : (8 + half * 4 + (c - 4)))];   // d += 32
        }
        cs[nb] = __builtin_amdgcn_wmma_f32_16x16x32_bf16(
            false, a0.v, false, blo.v, (short)0, cs[nb], false, false);
        cs[nb] = __builtin_amdgcn_wmma_f32_16x16x32_bf16(
            false, a1.v, false, bhi.v, (short)0, cs[nb], false, false);
      }

      unsigned short* pw = (unsigned short*)pbuf;
#pragma unroll
      for (int r = 0; r < 8; ++r) {           // online softmax, row = r + 8*half
        int qrow = qrow0 + r + half * 8;
        float s0 = cs[0][r] * 0.125f;         // 1/sqrt(64)
        float s1 = cs[1][r] * 0.125f;
        if (kt0 + lr > qrow) s0 = -1e30f;
        if (kt0 + 16 + lr > qrow) s1 = -1e30f;
        float tmax = fmaxf(s0, s1);
#pragma unroll
        for (int off = 8; off >= 1; off >>= 1) tmax = fmaxf(tmax, __shfl_xor(tmax, off, 32));
        float mnew = fmaxf(m_i[r], tmax);
        float p0 = __expf(s0 - mnew);
        float p1 = __expf(s1 - mnew);
        float corr = __expf(m_i[r] - mnew);
        float psum = p0 + p1;
#pragma unroll
        for (int off = 8; off >= 1; off >>= 1) psum += __shfl_xor(psum, off, 32);
        l_i[r] = l_i[r] * corr + psum;
        m_i[r] = mnew;
#pragma unroll
        for (int nblk = 0; nblk < 4; ++nblk) o[nblk][r] *= corr;
        int row = r + half * 8;
        pw[wave * 512 + row * 32 + lr]      = f2bf(p0);
        pw[wave * 512 + row * 32 + 16 + lr] = f2bf(p1);
      }
    }
    __syncthreads();                           // C-layout -> A-layout bounce through LDS
    if (active) {
      BFrag pf;
      const unsigned* pr = pbuf + wave * 256 + lr * 16;
#pragma unroll
      for (int c = 0; c < 8; ++c) {
        int idx = (c < 4) ? (half * 4 + c) : (8 + half * 4 + (c - 4));
        pf.u[c] = pr[idx];
      }
#pragma unroll
      for (int nblk = 0; nblk < 4; ++nblk) {   // o += P (16x32) . V (32x16-block)
        BFrag vf;
        int d = nblk * 16 + lr;
#pragma unroll
        for (int i = 0; i < 16; ++i) {
          int c = i >> 1, p = i & 1;
          int key = kt0 + ((c < 4) ? (half * 8 + c * 2 + p) : (16 + half * 8 + (c - 4) * 2 + p));
          vf.s[i] = v[((size_t)(b * SEQ + key) * HEADS + h) * HDIM + d];
        }
        o[nblk] = __builtin_amdgcn_wmma_f32_16x16x32_bf16(
            false, pf.v, false, vf.v, (short)0, o[nblk], false, false);
      }
    }
    __syncthreads();
  }

#pragma unroll
  for (int r = 0; r < 8; ++r) {
    float inv = 1.0f / l_i[r];
    int s = qrow0 + r + half * 8;
#pragma unroll
    for (int nblk = 0; nblk < 4; ++nblk) {
      int d = nblk * 16 + lr;
      ctx[((size_t)(b * SEQ + s) * HEADS + h) * HDIM + d] = f2bf(o[nblk][r] * inv);
    }
  }
}

// ---------------------------------------------------------------- launcher
extern "C" void kernel_launch(void* const* d_in, const int* in_sizes, int n_in,
                              void* d_out, int out_size, void* d_ws, size_t ws_size,
                              hipStream_t stream) {
  (void)in_sizes; (void)n_in; (void)out_size; (void)ws_size;
  const float* x  = (const float*)d_in[0];
  const float* wq = (const float*)d_in[1];
  const float* wk = (const float*)d_in[2];
  const float* wv = (const float*)d_in[3];
  const float* wo = (const float*)d_in[4];
  const float* bo = (const float*)d_in[5];
  const float* w1 = (const float*)d_in[6];
  const float* b1 = (const float*)d_in[7];
  const float* w2 = (const float*)d_in[8];
  const float* b2 = (const float*)d_in[9];
  const float* g1 = (const float*)d_in[10];
  const float* s1 = (const float*)d_in[11];
  const float* g2 = (const float*)d_in[12];
  const float* s2 = (const float*)d_in[13];
  float* out = (float*)d_out;

  char* ws = (char*)d_ws;
  size_t off = 0;
  auto alloc = [&](size_t elems) {
    unsigned short* p = (unsigned short*)(ws + off);
    off += ((elems * 2 + 255) & ~(size_t)255);
    return p;
  };
  unsigned short* wq_t  = alloc((size_t)EMB * EMB);   // transposed bf16 weights (N x K)
  unsigned short* wk_t  = alloc((size_t)EMB * EMB);
  unsigned short* wv_t  = alloc((size_t)EMB * EMB);
  unsigned short* wo_t  = alloc((size_t)EMB * EMB);
  unsigned short* w1_t  = alloc((size_t)EMB * FF);    // FF x EMB
  unsigned short* w2_t  = alloc((size_t)FF * EMB);    // EMB x FF
  unsigned short* h_b   = alloc((size_t)MTOT * EMB);  // LN1 out, reused for LN2 out
  unsigned short* q_b   = alloc((size_t)MTOT * EMB);
  unsigned short* k_b   = alloc((size_t)MTOT * EMB);
  unsigned short* v_b   = alloc((size_t)MTOT * EMB);
  unsigned short* ctx_b = alloc((size_t)MTOT * EMB);
  unsigned short* f1_b  = alloc((size_t)MTOT * FF);

  // 1) weights fp32 -> bf16, transposed to (N x K) for TDM-friendly row-major tiles
  dim3 t256(256);
  cvt_transpose_bf16<<<dim3(EMB / 32, EMB / 32), t256, 0, stream>>>(wq, wq_t, EMB, EMB);
  cvt_transpose_bf16<<<dim3(EMB / 32, EMB / 32), t256, 0, stream>>>(wk, wk_t, EMB, EMB);
  cvt_transpose_bf16<<<dim3(EMB / 32, EMB / 32), t256, 0, stream>>>(wv, wv_t, EMB, EMB);
  cvt_transpose_bf16<<<dim3(EMB / 32, EMB / 32), t256, 0, stream>>>(wo, wo_t, EMB, EMB);
  cvt_transpose_bf16<<<dim3(FF / 32, EMB / 32),  t256, 0, stream>>>(w1, w1_t, EMB, FF);
  cvt_transpose_bf16<<<dim3(EMB / 32, FF / 32),  t256, 0, stream>>>(w2, w2_t, FF, EMB);

  // 2) LN1
  layernorm_bf16<<<MTOT, 256, 0, stream>>>(x, g1, s1, h_b);

  // 3) Q,K,V projections
  dim3 gproj(EMB / 128, MTOT / 128);
  wmma_gemm_bf16<true, false><<<gproj, 128, 0, stream>>>(h_b, wq_t, nullptr, nullptr, q_b, MTOT, EMB, EMB);
  wmma_gemm_bf16<true, false><<<gproj, 128, 0, stream>>>(h_b, wk_t, nullptr, nullptr, k_b, MTOT, EMB, EMB);
  wmma_gemm_bf16<true, false><<<gproj, 128, 0, stream>>>(h_b, wv_t, nullptr, nullptr, v_b, MTOT, EMB, EMB);

  // 4) causal flash attention
  flash_attn_wmma<<<dim3(BATCH * HEADS, SEQ / 64), 128, 0, stream>>>(q_b, k_b, v_b, ctx_b);

  // 5) output projection + bias + residual(x) -> fp32 d_out
  wmma_gemm_bf16<false, false><<<gproj, 128, 0, stream>>>(ctx_b, wo_t, bo, x, out, MTOT, EMB, EMB);

  // 6) LN2
  layernorm_bf16<<<MTOT, 256, 0, stream>>>(out, g2, s2, h_b);

  // 7) FFN up + GELU -> bf16
  wmma_gemm_bf16<true, true><<<dim3(FF / 128, MTOT / 128), 128, 0, stream>>>(h_b, w1_t, b1, nullptr, f1_b, MTOT, FF, EMB);

  // 8) FFN down + bias + residual(d_out) -> fp32 d_out
  wmma_gemm_bf16<false, false><<<gproj, 128, 0, stream>>>(f1_b, w2_t, b2, out, out, MTOT, EMB, FF);
}
